// AttentionBlock_68564857914229
// MI455X (gfx1250) — matrix operand order repaired and checked by reading, not compiled, not run on hardware
//
#include <hip/hip_runtime.h>

// ---------------------------------------------------------------------------
// AttentionBlock on gfx1250 (MI455X): GroupNorm -> QKV GEMM -> QK^T+softmax ->
// attn*V -> proj GEMM (+bias+residual).  All GEMMs via v_wmma_f32_16x16x32_f16
// (f16 inputs, f32 accumulate).  Intermediates (~114 MB f16) live in d_ws and
// fit in the 192MB L2.  GEMM waves compute 64x64 tiles (4x4 WMMA accumulators)
// so each K=32 step issues 16 WMMAs against 16 b128 loads (~32 FLOP per loaded
// byte) -- WMMA issue-rate bound, not load bound.
// ---------------------------------------------------------------------------

typedef __attribute__((ext_vector_type(16))) _Float16 v16h;
typedef __attribute__((ext_vector_type(8)))  _Float16 v8h;
typedef __attribute__((ext_vector_type(8)))  float    v8f;

static constexpr int BATCH = 16;
static constexpr int CH    = 512;   // channels
static constexpr int NPIX  = 1024;  // H*W

// ---------------- WMMA helpers ----------------------------------------------

__device__ __forceinline__ v16h cat8(v8h lo, v8h hi) {
  return __builtin_shufflevector(lo, hi, 0,1,2,3,4,5,6,7,8,9,10,11,12,13,14,15);
}

// A fragment (16x32 f16, M x K), source row-major with leading dim lda.
// Lanes 0-15: row M=lane, K = 0..15; lanes 16-31: K = 16..31.
__device__ __forceinline__ v16h load_a(const _Float16* __restrict__ A, long lda,
                                       int k0, int lane) {
  const int half = lane >> 4, m = lane & 15;
  const _Float16* p = A + (long)m * lda + k0 + half * 16;
  return cat8(*(const v8h*)p, *(const v8h*)(p + 8));
}

// B fragment (32x16 f16, K x N), source stored column-major (K contiguous),
// leading dim ldb.  Lane holds column N=lane&15, K = (lane>>4)*16 .. +15.
__device__ __forceinline__ v16h load_b(const _Float16* __restrict__ Bc, long ldb,
                                       int k0, int lane) {
  const int half = lane >> 4, n = lane & 15;
  const _Float16* p = Bc + (long)n * ldb + k0 + half * 16;
  return *(const v16h*)p; // 32B contiguous
}

__device__ __forceinline__ v8f wmma16(v16h a, v16h b, v8f c) {
  return __builtin_amdgcn_wmma_f32_16x16x32_f16(false, a, false, b,
                                                (short)0, c, false, false);
}

// 64(M) x 64(N) wave tile: 4x4 accumulators, 16 WMMA per K=32 step.
template <int KTOT>
__device__ __forceinline__ void gemm_wave_tile(const _Float16* __restrict__ A, long lda,
                                               const _Float16* __restrict__ Bc, long ldb,
                                               int lane, v8f acc[4][4]) {
  for (int k0 = 0; k0 < KTOT; k0 += 32) {
    v16h a[4], b[4];
#pragma unroll
    for (int i = 0; i < 4; ++i) a[i] = load_a(A + (long)16 * i * lda, lda, k0, lane);
#pragma unroll
    for (int j = 0; j < 4; ++j) b[j] = load_b(Bc + (long)16 * j * ldb, ldb, k0, lane);
#pragma unroll
    for (int i = 0; i < 4; ++i)
#pragma unroll
      for (int j = 0; j < 4; ++j)
        acc[i][j] = wmma16(a[i], b[j], acc[i][j]);
  }
}

// ---------------- Kernel 0: fp32 -> f16 weight conversion --------------------

__global__ void cvt_f16_kernel(const float* __restrict__ src,
                               _Float16* __restrict__ dst, int n) {
  int i = blockIdx.x * blockDim.x + threadIdx.x;
  if (i < n) dst[i] = (_Float16)src[i];
}

// ---------------- Kernel 1: GroupNorm -> h^T [b][n][c] f16 -------------------

__global__ void __launch_bounds__(256)
gn_kernel(const float* __restrict__ x, const float* __restrict__ scale,
          const float* __restrict__ bias, _Float16* __restrict__ h_t) {
  const int b = blockIdx.x >> 5;   // 16 batches
  const int g = blockIdx.x & 31;   // 32 groups, 16 ch each
  const long base = ((long)b * CH + g * 16) * NPIX;
  float s = 0.f, s2 = 0.f;
  for (int i = threadIdx.x; i < 16 * NPIX; i += 256) {
    float v = x[base + i];
    s += v; s2 += v * v;
  }
#pragma unroll
  for (int off = 16; off; off >>= 1) {
    s  += __shfl_xor(s, off, 32);
    s2 += __shfl_xor(s2, off, 32);
  }
  __shared__ float rs[8], rs2[8];
  const int lane = threadIdx.x & 31, wave = threadIdx.x >> 5;
  if (lane == 0) { rs[wave] = s; rs2[wave] = s2; }
  __syncthreads();
  if (wave == 0) {
    s  = (lane < 8) ? rs[lane]  : 0.f;
    s2 = (lane < 8) ? rs2[lane] : 0.f;
#pragma unroll
    for (int off = 4; off; off >>= 1) {
      s  += __shfl_xor(s, off, 32);
      s2 += __shfl_xor(s2, off, 32);
    }
    if (lane == 0) { rs[0] = s; rs2[0] = s2; }
  }
  __syncthreads();
  const float mean = rs[0] * (1.f / 16384.f);
  const float var  = rs2[0] * (1.f / 16384.f) - mean * mean;
  const float rstd = __frsqrt_rn(var + 1e-6f);
  for (int i = threadIdx.x; i < 16 * NPIX; i += 256) {
    const int ch = i >> 10, n = i & 1023;
    const int c  = g * 16 + ch;
    const float v = (x[base + i] - mean) * rstd * scale[c] + bias[c];
    h_t[((long)b * NPIX + n) * CH + c] = (_Float16)v;
  }
}

// ---------------- Kernel 2: QKV GEMM (per batch 1536x1024, K=512) ------------
// A = W_qkv (1536x512 row-major f16), B = h^T (col-major over c).
// Writes q^T,k^T as [b][n][c] and v as [b][c][n].  Block tile 128x256.

__global__ void __launch_bounds__(256)
qkv_gemm_kernel(const _Float16* __restrict__ Wq, const _Float16* __restrict__ h_t,
                const float* __restrict__ qkv_b,
                _Float16* __restrict__ q_t, _Float16* __restrict__ k_t,
                _Float16* __restrict__ v) {
  const int lane = threadIdx.x & 31;
  const int wave = threadIdx.x >> 5;
  const int mBase = blockIdx.x * 128 + (wave & 1) * 64;  // output row o
  const int nBase = blockIdx.y * 256 + (wave >> 1) * 64; // pixel n
  const int b = blockIdx.z;
  const _Float16* A  = Wq  + (long)mBase * CH;
  const _Float16* Bc = h_t + ((long)b * NPIX + nBase) * CH;
  v8f acc[4][4] = {};
  gemm_wave_tile<CH>(A, CH, Bc, CH, lane, acc);
  const int half = lane >> 4, nl = lane & 15;
#pragma unroll
  for (int i = 0; i < 4; ++i)
#pragma unroll
    for (int j = 0; j < 4; ++j)
#pragma unroll
      for (int r = 0; r < 8; ++r) {
        const int o = mBase + i * 16 + r + half * 8;
        const int n = nBase + j * 16 + nl;
        const float val = acc[i][j][r] + qkv_b[o];
        if (o < CH)            q_t[((long)b * NPIX + n) * CH + o]          = (_Float16)val;
        else if (o < 2 * CH)   k_t[((long)b * NPIX + n) * CH + (o - CH)]   = (_Float16)val;
        else                   v[((long)b * CH + (o - 2 * CH)) * NPIX + n] = (_Float16)val;
      }
}

// ---------------- Kernel 3: S = scale*q^T k, softmax, -> attn f16 ------------
// Block = 16 query rows x all 1024 keys; q A-frags held in 128 VGPRs, k
// streamed; 64KB LDS row buffer for the f32 softmax.

__global__ void __launch_bounds__(256)
scores_softmax_kernel(const _Float16* __restrict__ q_t,
                      const _Float16* __restrict__ k_t,
                      _Float16* __restrict__ attn) {
  __shared__ float S[16 * NPIX]; // 64 KB
  const int lane = threadIdx.x & 31;
  const int wave = threadIdx.x >> 5;
  const int nTile = blockIdx.x, b = blockIdx.y;
  const _Float16* qA = q_t + ((long)b * NPIX + nTile * 16) * CH;
  v16h a[16];
#pragma unroll
  for (int kc = 0; kc < 16; ++kc) a[kc] = load_a(qA, CH, kc * 32, lane);
  const float scl = 0.044194173824159216f; // 512^-0.5
  const int half = lane >> 4, nl = lane & 15;
  for (int mt = wave * 8; mt < wave * 8 + 8; ++mt) {
    const _Float16* kB = k_t + ((long)b * NPIX + mt * 16) * CH;
    v8f acc = {};
#pragma unroll
    for (int kc = 0; kc < 16; ++kc)
      acc = wmma16(a[kc], load_b(kB, CH, kc * 32, lane), acc);
#pragma unroll
    for (int r = 0; r < 8; ++r)
      S[(r + half * 8) * NPIX + mt * 16 + nl] = acc[r] * scl;
  }
  __syncthreads();
  // softmax: each wave owns 2 rows, 32 cols/lane
#pragma unroll
  for (int rr = 0; rr < 2; ++rr) {
    const int row = wave * 2 + rr;
    const float* Sr = S + row * NPIX;
    float mx = -3.4e38f;
#pragma unroll
    for (int i = 0; i < 32; ++i) mx = fmaxf(mx, Sr[lane + i * 32]);
#pragma unroll
    for (int off = 16; off; off >>= 1) mx = fmaxf(mx, __shfl_xor(mx, off, 32));
    float e[32], sum = 0.f;
#pragma unroll
    for (int i = 0; i < 32; ++i) { e[i] = __expf(Sr[lane + i * 32] - mx); sum += e[i]; }
#pragma unroll
    for (int off = 16; off; off >>= 1) sum += __shfl_xor(sum, off, 32);
    const float inv = 1.0f / sum;
    _Float16* arow = attn + ((long)b * NPIX + nTile * 16 + row) * NPIX;
#pragma unroll
    for (int i = 0; i < 32; ++i) arow[lane + i * 32] = (_Float16)(e[i] * inv);
  }
}

// ---------------- Kernel 4: out^T = attn @ v^T  (M=n, N=c, K=m=1024) ---------

__global__ void __launch_bounds__(256)
av_gemm_kernel(const _Float16* __restrict__ attn, const _Float16* __restrict__ v,
               _Float16* __restrict__ out_t) {
  const int lane = threadIdx.x & 31;
  const int wave = threadIdx.x >> 5;
  const int mBase = blockIdx.x * 128 + (wave & 1) * 64;  // n
  const int nBase = blockIdx.y * 256 + (wave >> 1) * 64; // c
  const int b = blockIdx.z;
  const _Float16* A  = attn + ((long)b * NPIX + mBase) * NPIX;
  const _Float16* Bc = v    + ((long)b * CH + nBase) * NPIX;
  v8f acc[4][4] = {};
  gemm_wave_tile<NPIX>(A, NPIX, Bc, NPIX, lane, acc);
  const int half = lane >> 4, nl = lane & 15;
#pragma unroll
  for (int i = 0; i < 4; ++i)
#pragma unroll
    for (int j = 0; j < 4; ++j)
#pragma unroll
      for (int r = 0; r < 8; ++r) {
        const int n = mBase + i * 16 + r + half * 8;
        const int c = nBase + j * 16 + nl;
        out_t[((long)b * NPIX + n) * CH + c] = (_Float16)acc[i][j][r];
      }
}

// ---------------- Kernel 5: proj GEMM + bias + residual (fp32 out) -----------

__global__ void __launch_bounds__(256)
proj_gemm_kernel(const _Float16* __restrict__ Pw, const _Float16* __restrict__ out_t,
                 const float* __restrict__ proj_b, const float* __restrict__ x,
                 float* __restrict__ y) {
  const int lane = threadIdx.x & 31;
  const int wave = threadIdx.x >> 5;
  const int mBase = blockIdx.x * 128 + (wave & 1) * 64;  // o
  const int nBase = blockIdx.y * 256 + (wave >> 1) * 64; // n
  const int b = blockIdx.z;
  const _Float16* A  = Pw    + (long)mBase * CH;
  const _Float16* Bc = out_t + ((long)b * NPIX + nBase) * CH;
  v8f acc[4][4] = {};
  gemm_wave_tile<CH>(A, CH, Bc, CH, lane, acc);
  const int half = lane >> 4, nl = lane & 15;
#pragma unroll
  for (int i = 0; i < 4; ++i)
#pragma unroll
    for (int j = 0; j < 4; ++j)
#pragma unroll
      for (int r = 0; r < 8; ++r) {
        const int o = mBase + i * 16 + r + half * 8;
        const int n = nBase + j * 16 + nl;
        const long idx = ((long)b * CH + o) * NPIX + n;
        y[idx] = acc[i][j][r] + proj_b[o] + x[idx];
      }
}

// ---------------- Launcher ---------------------------------------------------

extern "C" void kernel_launch(void* const* d_in, const int* in_sizes, int n_in,
                              void* d_out, int out_size, void* d_ws, size_t ws_size,
                              hipStream_t stream) {
  (void)in_sizes; (void)n_in; (void)out_size; (void)ws_size;
  const float* x        = (const float*)d_in[0];
  const float* gn_scale = (const float*)d_in[1];
  const float* gn_bias  = (const float*)d_in[2];
  const float* qkv_w    = (const float*)d_in[3];
  const float* qkv_b    = (const float*)d_in[4];
  const float* proj_w   = (const float*)d_in[5];
  const float* proj_b   = (const float*)d_in[6];
  float* y = (float*)d_out;

  // Workspace carve-up (f16, 256B aligned); total ~114 MiB.
  char* ws = (char*)d_ws;
  size_t off = 0;
  auto carve = [&](size_t elems) {
    void* p = ws + off;
    off += (elems * sizeof(_Float16) + 255) & ~(size_t)255;
    return (_Float16*)p;
  };
  const size_t BNC = (size_t)BATCH * NPIX * CH;     // 8M elems
  _Float16* h_t   = carve(BNC);
  _Float16* q_t   = carve(BNC);
  _Float16* k_t   = carve(BNC);
  _Float16* v     = carve(BNC);
  _Float16* attn  = carve((size_t)BATCH * NPIX * NPIX);
  _Float16* out_t = carve(BNC);
  _Float16* wq16  = carve((size_t)3 * CH * CH);
  _Float16* wp16  = carve((size_t)CH * CH);

  const int nWq = 3 * CH * CH, nWp = CH * CH;
  cvt_f16_kernel<<<dim3((nWq + 255) / 256), 256, 0, stream>>>(qkv_w, wq16, nWq);
  cvt_f16_kernel<<<dim3((nWp + 255) / 256), 256, 0, stream>>>(proj_w, wp16, nWp);
  gn_kernel<<<dim3(BATCH * 32), 256, 0, stream>>>(x, gn_scale, gn_bias, h_t);
  qkv_gemm_kernel<<<dim3(12, 4, BATCH), 256, 0, stream>>>(wq16, h_t, qkv_b, q_t, k_t, v);
  scores_softmax_kernel<<<dim3(64, BATCH), 256, 0, stream>>>(q_t, k_t, attn);
  av_gemm_kernel<<<dim3(8, 2, BATCH), 256, 0, stream>>>(attn, v, out_t);
  proj_gemm_kernel<<<dim3(4, 4, BATCH), 256, 0, stream>>>(wp16, out_t, proj_b, x, y);
}
